// RNN_11287174054562
// MI455X (gfx1250) — matrix operand order repaired and checked by reading, not compiled, not run on hardware
//
#include <hip/hip_runtime.h>
#include <hip/hip_bf16.h>
#include <math.h>

// Problem dims
#define S_ 64
#define B_ 32
#define H_ 1024
#define E_ 512
#define V_ 32000
#define L_ 2

typedef __bf16 bf16x16 __attribute__((ext_vector_type(16)));
typedef float  f32x8   __attribute__((ext_vector_type(8)));

union Frag {
  bf16x16 v;
  uint4   q[2];
};

__device__ __forceinline__ unsigned short f32_to_bf16(float f) {
  unsigned u = __float_as_uint(f);
  u += 0x7FFFu + ((u >> 16) & 1u);   // round-to-nearest-even
  return (unsigned short)(u >> 16);
}

// ---------------------------------------------------------- async-to-LDS path
// CDNA5 GLOBAL_LOAD_ASYNC_TO_LDS_B128: DMA 16B/lane global->LDS, ASYNCcnt.
// Probe round 2: builtin exists with signature
//   (int4 addrspace(1)*, int4 addrspace(3)*, imm offset, imm cpol)
typedef int v4i_ __attribute__((ext_vector_type(4)));
typedef __attribute__((address_space(1))) v4i_* as1_p4;
typedef __attribute__((address_space(3))) v4i_* as3_p4;

__device__ __forceinline__ unsigned lds_offset_of(const void* p) {
  return (unsigned)(unsigned long long)
      (const __attribute__((address_space(3))) void*)p;
}

#if __has_builtin(__builtin_amdgcn_global_load_async_to_lds_b128)
#define ASYNC_B128(gp, lp, OFF, OFFSTR)                                       \
  __builtin_amdgcn_global_load_async_to_lds_b128((as1_p4)(gp), (as3_p4)(lp),  \
                                                 OFF, 0)
#else
// fallback: inline asm per CDNA5 ISA (vdst=LDS addr VGPR,
// vaddr=64-bit global addr; offset applies to both sides)
#define ASYNC_B128(gp, lp, OFF, OFFSTR)                                       \
  asm volatile("global_load_async_to_lds_b128 %0, %1, off offset:" OFFSTR     \
               :: "v"(lds_offset_of(lp)), "v"(gp) : "memory")
#endif

#if __has_builtin(__builtin_amdgcn_s_wait_asynccnt)
#define WAIT_ASYNC(N, NSTR) __builtin_amdgcn_s_wait_asynccnt(N)
#else
#define WAIT_ASYNC(N, NSTR) \
  asm volatile("s_wait_asynccnt " NSTR ::: "memory")
#endif

// ---------------------------------------------------------------- converters
__global__ void k_cvt_bf16(const float* __restrict__ src,
                           unsigned short* __restrict__ dst, int n) {
  int i = blockIdx.x * blockDim.x + threadIdx.x;
  int stride = gridDim.x * blockDim.x;
  for (; i < n; i += stride) dst[i] = f32_to_bf16(src[i]);
}

__global__ void k_embed(const int* __restrict__ tok,
                        const float* __restrict__ emb,
                        unsigned short* __restrict__ xb) {
  int i = blockIdx.x * blockDim.x + threadIdx.x;      // over S*B*E
  if (i >= S_ * B_ * E_) return;
  int e  = i & (E_ - 1);
  int sb = i >> 9;                                    // E_ == 512
  int row = tok[sb];
  xb[i] = f32_to_bf16(emb[(size_t)row * E_ + e]);
}

__global__ void k_zero_barrier(unsigned* b) {
  if (threadIdx.x < 2) b[threadIdx.x] = 0u;
}

// ---------------------------------------------------------------- grid barrier
__device__ __forceinline__ void grid_barrier(unsigned* cnt, unsigned* gen,
                                             unsigned nwg) {
  __builtin_amdgcn_fence(__ATOMIC_RELEASE, "agent");
  __syncthreads();
  if (threadIdx.x == 0) {
    unsigned g = __hip_atomic_load(gen, __ATOMIC_RELAXED, __HIP_MEMORY_SCOPE_AGENT);
    unsigned a = __hip_atomic_fetch_add(cnt, 1u, __ATOMIC_ACQ_REL, __HIP_MEMORY_SCOPE_AGENT);
    if (a == nwg - 1u) {
      __hip_atomic_store(cnt, 0u, __ATOMIC_RELAXED, __HIP_MEMORY_SCOPE_AGENT);
      __hip_atomic_fetch_add(gen, 1u, __ATOMIC_RELEASE, __HIP_MEMORY_SCOPE_AGENT);
    } else {
      while (__hip_atomic_load(gen, __ATOMIC_ACQUIRE, __HIP_MEMORY_SCOPE_AGENT) == g) {
        __builtin_amdgcn_s_sleep(1);
      }
    }
  }
  __syncthreads();
  __builtin_amdgcn_fence(__ATOMIC_ACQUIRE, "agent");
}

// ------------------------------------------------- WMMA K-span accumulation
// A: rows (lane&15), K-chunks {kb..kb+7, kb+16..kb+23} (+8 for hi lanes)
// W: row-major [N][K]; B column n=(lane&15) uses same K-chunks as A from W row n
__device__ __forceinline__ f32x8 wmma_span(const unsigned short* __restrict__ A, int lda,
                                           const unsigned short* __restrict__ W, int ldw,
                                           int K, f32x8 c) {
  const int lane = threadIdx.x & 31;
  const unsigned short* arow = A + (lane & 15) * lda + ((lane >> 4) << 3);
  const unsigned short* wrow = W + (lane & 15) * ldw + ((lane >> 4) << 3);
  for (int kb = 0; kb < K; kb += 32) {
    Frag a, b;
    a.q[0] = *(const uint4*)(arow + kb);
    a.q[1] = *(const uint4*)(arow + kb + 16);
    b.q[0] = *(const uint4*)(wrow + kb);
    b.q[1] = *(const uint4*)(wrow + kb + 16);
    c = __builtin_amdgcn_wmma_f32_16x16x32_bf16(false, a.v, false, b.v,
                                                (short)0, c, false, false);
  }
  return c;
}

// ---------------------------------------------------------------- RNN kernel
#define RNN_NWG 32
#define RNN_THREADS 128   // 4 waves; 32*4 = 128 waves = 2(M) x 64(N) tiles

__global__ __launch_bounds__(RNN_THREADS, 1)
void k_rnn(const unsigned short* __restrict__ xb,   // [S][B][E] bf16
           const unsigned short* __restrict__ wx0,  // [H][E]
           const unsigned short* __restrict__ wh0,  // [H][H]
           const unsigned short* __restrict__ wx1,  // [H][H]
           const unsigned short* __restrict__ wh1,  // [H][H]
           const float* __restrict__ bh0,
           const float* __restrict__ bh1,
           const float* __restrict__ hidden_in,     // [L][B][H] f32
           unsigned short* __restrict__ hb,         // [L][2][B*H] bf16 state
           float* __restrict__ hf,                  // [L][2][B*H] f32 state
           unsigned short* __restrict__ tops,       // [S][B*H] bf16
           float* __restrict__ hid_out,             // d_out + S*B*V
           unsigned* cnt, unsigned* gen)
{
  const int tid = blockIdx.x * blockDim.x + threadIdx.x;
  const int BH  = B_ * H_;

  // Load initial hidden state into buffer 0 (f32 + bf16 mirrors)
  for (int i = tid; i < L_ * BH; i += RNN_NWG * RNN_THREADS) {
    int l = i / BH, j = i - l * BH;
    float v = hidden_in[i];
    hf[(l * 2 + 0) * BH + j] = v;
    hb[(l * 2 + 0) * BH + j] = f32_to_bf16(v);
  }
  grid_barrier(cnt, gen, RNN_NWG);

  const int wid   = (blockIdx.x << 2) + (threadIdx.x >> 5);  // 0..127
  const int lane  = threadIdx.x & 31;
  const int mBase = (wid & 1) << 4;                          // batch tile
  const int nBase = (wid >> 1) << 4;                         // hidden tile
  const int col   = nBase + (lane & 15);
  const int rb0   = mBase + ((lane >> 4) << 3);

  int cur = 0;
  for (int t = 0; t < S_; ++t) {
    const int nxt = cur ^ 1;
    // ---------- layer 0: h0' = tanh(x_t Wx0^T + h0 Wh0^T + bh0)
    {
      f32x8 c = {};
      c = wmma_span(xb + t * (B_ * E_) + mBase * E_, E_, wx0 + nBase * E_, E_, E_, c);
      c = wmma_span(hb + (0 * 2 + cur) * BH + mBase * H_, H_, wh0 + nBase * H_, H_, H_, c);
      const float bv = bh0[col];
      float*          hfw = hf + (0 * 2 + nxt) * BH;
      unsigned short* hbw = hb + (0 * 2 + nxt) * BH;
#pragma unroll
      for (int r = 0; r < 8; ++r) {
        float v = tanhf(c[r] + bv);
        hfw[(rb0 + r) * H_ + col] = v;
        hbw[(rb0 + r) * H_ + col] = f32_to_bf16(v);
      }
    }
    grid_barrier(cnt, gen, RNN_NWG);
    // ---------- layer 1: h1' = tanh(h0' Wx1^T + h1 Wh1^T + bh1)
    {
      f32x8 c = {};
      c = wmma_span(hb + (0 * 2 + nxt) * BH + mBase * H_, H_, wx1 + nBase * H_, H_, H_, c);
      c = wmma_span(hb + (1 * 2 + cur) * BH + mBase * H_, H_, wh1 + nBase * H_, H_, H_, c);
      const float bv = bh1[col];
      float*          hfw = hf + (1 * 2 + nxt) * BH;
      unsigned short* hbw = hb + (1 * 2 + nxt) * BH;
      unsigned short* tw  = tops + t * BH;
#pragma unroll
      for (int r = 0; r < 8; ++r) {
        float v = tanhf(c[r] + bv);
        hfw[(rb0 + r) * H_ + col] = v;
        unsigned short u = f32_to_bf16(v);
        hbw[(rb0 + r) * H_ + col] = u;
        tw [(rb0 + r) * H_ + col] = u;
      }
    }
    grid_barrier(cnt, gen, RNN_NWG);
    cur = nxt;
  }

  // Final hidden -> d_out tail ([L][B][H], f32), visible after last barrier
  for (int i = tid; i < L_ * BH; i += RNN_NWG * RNN_THREADS) {
    int l = i / BH, j = i - l * BH;
    hid_out[i] = hf[(l * 2 + cur) * BH + j];
  }
}

// ------------------------------------------------------------- projection GEMM
// logits[m][n] = sum_k tops[m][k] * Wout[n][k] + bout[n]
// M = S*B = 2048, N = V = 32000, K = H = 1024
// Wout bf16 (64MB) fits in the 192MB L2; logits write (262MB) is the roofline
// floor, so the GEMM only needs enough overlap to hide L2 latency: the Wout
// tile is DMA'd straight into LDS with CDNA5 async-to-LDS ops, double-buffered
// against the WMMA inner loop (ASYNCcnt completes in order: after issuing the
// next buffer's 4 transfers, waiting ASYNCcnt<=4 proves the current buffer's
// 4 transfers are done).
#define PJ_THREADS 256
#define PJ_MBLK 64
#define PJ_NBLK 256
#define LDSB_STRIDE 40   // shorts (32 data + 8 pad) -> 80B rows, spreads banks

__global__ __launch_bounds__(PJ_THREADS, 2)
void k_proj(const unsigned short* __restrict__ A,   // tops [2048][1024] bf16
            const unsigned short* __restrict__ W,   // wout [32000][1024] bf16
            const float* __restrict__ bout,
            float* __restrict__ out)                // logits [2048][32000] f32
{
  __shared__ unsigned short Bs[2][PJ_NBLK * LDSB_STRIDE];  // 2 x 20 KB
  const int nBlock = blockIdx.x * PJ_NBLK;
  const int mBlock = blockIdx.y * PJ_MBLK;
  const int w     = threadIdx.x >> 5;
  const int lane  = threadIdx.x & 31;
  const int mSub  = w & 3;          // 4 M-subtiles of 16
  const int nHalf = w >> 2;         // 2 N-halves of 128
  const unsigned short* aPtr =
      A + (size_t)(mBlock + (mSub << 4) + (lane & 15)) * H_ + ((lane >> 4) << 3);
  const unsigned short* wRow = W + (size_t)(nBlock + threadIdx.x) * H_;

  f32x8 acc[8] = {};

  // prologue: stage K-step 0 into buffer 0 (64B/thread = 4 async b128)
  {
    const unsigned short* g = wRow;
    unsigned short* l = &Bs[0][threadIdx.x * LDSB_STRIDE];
    ASYNC_B128(g, l, 0,  "0");
    ASYNC_B128(g, l, 16, "16");
    ASYNC_B128(g, l, 32, "32");
    ASYNC_B128(g, l, 48, "48");
  }

  for (int kb = 0, it = 0; kb < H_; kb += 32, ++it) {
    const int cur = it & 1;
    if (kb + 32 < H_) {
      // prefetch next K-step into the other buffer, then wait for current
      const unsigned short* g = wRow + kb + 32;
      unsigned short* l = &Bs[cur ^ 1][threadIdx.x * LDSB_STRIDE];
      ASYNC_B128(g, l, 0,  "0");
      ASYNC_B128(g, l, 16, "16");
      ASYNC_B128(g, l, 32, "32");
      ASYNC_B128(g, l, 48, "48");
      WAIT_ASYNC(4, "0x4");   // in-order completion => current 4 are done
    } else {
      WAIT_ASYNC(0, "0x0");
    }
    __syncthreads();          // all waves' current-buffer DMA complete

    Frag a;
    a.q[0] = *(const uint4*)(aPtr + kb);
    a.q[1] = *(const uint4*)(aPtr + kb + 16);
#pragma unroll
    for (int j = 0; j < 8; ++j) {
      const unsigned short* ls =
          &Bs[cur][((nHalf << 7) + (j << 4) + (lane & 15)) * LDSB_STRIDE +
                   ((lane >> 4) << 3)];
      Frag b;
      b.q[0] = *(const uint4*)(ls);
      b.q[1] = *(const uint4*)(ls + 16);
      acc[j] = __builtin_amdgcn_wmma_f32_16x16x32_bf16(false, a.v, false, b.v,
                                                       (short)0, acc[j], false, false);
    }
    __syncthreads();          // readers done before this buffer is re-staged
  }

  const int rb = mBlock + (mSub << 4) + ((lane >> 4) << 3);
#pragma unroll
  for (int j = 0; j < 8; ++j) {
    const int ncol = nBlock + (nHalf << 7) + (j << 4) + (lane & 15);
    const float bv = bout[ncol];
#pragma unroll
    for (int r = 0; r < 8; ++r) {
      out[(size_t)(rb + r) * V_ + ncol] = acc[j][r] + bv;
    }
  }
}

// ------------------------------------------------------------------- launcher
extern "C" void kernel_launch(void* const* d_in, const int* in_sizes, int n_in,
                              void* d_out, int out_size, void* d_ws, size_t ws_size,
                              hipStream_t stream) {
  (void)in_sizes; (void)n_in; (void)out_size; (void)ws_size;

  const int*   tok    = (const int*)  d_in[0];
  const float* hidden = (const float*)d_in[1];
  const float* emb    = (const float*)d_in[2];
  const float* Wx0    = (const float*)d_in[3];
  const float* Wh0    = (const float*)d_in[4];
  const float* bh0    = (const float*)d_in[5];
  const float* Wx1    = (const float*)d_in[6];
  const float* Wh1    = (const float*)d_in[7];
  const float* bh1    = (const float*)d_in[8];
  const float* Wout   = (const float*)d_in[9];
  const float* bout   = (const float*)d_in[10];
  float* out = (float*)d_out;

  // Workspace carve-up (~79 MB total)
  char* p = (char*)d_ws;
  auto carve = [&](size_t bytes) -> char* {
    char* r = p; p += (bytes + 255) & ~(size_t)255; return r;
  };
  unsigned short* woutB = (unsigned short*)carve((size_t)V_ * H_ * 2);
  unsigned short* wx0B  = (unsigned short*)carve((size_t)H_ * E_ * 2);
  unsigned short* wh0B  = (unsigned short*)carve((size_t)H_ * H_ * 2);
  unsigned short* wx1B  = (unsigned short*)carve((size_t)H_ * H_ * 2);
  unsigned short* wh1B  = (unsigned short*)carve((size_t)H_ * H_ * 2);
  unsigned short* xB    = (unsigned short*)carve((size_t)S_ * B_ * E_ * 2);
  unsigned short* topsB = (unsigned short*)carve((size_t)S_ * B_ * H_ * 2);
  unsigned short* hB    = (unsigned short*)carve((size_t)L_ * 2 * B_ * H_ * 2);
  float*          hF    = (float*)         carve((size_t)L_ * 2 * B_ * H_ * 4);
  unsigned*       bar   = (unsigned*)      carve(256);

  float* hid_out = out + (size_t)S_ * B_ * V_;

  // 1) precision conversion (Wout 64MB bf16 -> fits MI455X 192MB L2)
  k_cvt_bf16<<<2048, 256, 0, stream>>>(Wout, woutB, V_ * H_);
  k_cvt_bf16<<<512,  256, 0, stream>>>(Wx0,  wx0B,  H_ * E_);
  k_cvt_bf16<<<512,  256, 0, stream>>>(Wh0,  wh0B,  H_ * H_);
  k_cvt_bf16<<<512,  256, 0, stream>>>(Wx1,  wx1B,  H_ * H_);
  k_cvt_bf16<<<512,  256, 0, stream>>>(Wh1,  wh1B,  H_ * H_);

  // 2) embedding gather -> bf16
  k_embed<<<(S_ * B_ * E_ + 255) / 256, 256, 0, stream>>>(tok, emb, xB);

  // 3) recurrent part: persistent cooperative kernel, 2 grid barriers / step
  k_zero_barrier<<<1, 32, 0, stream>>>(bar);
  k_rnn<<<RNN_NWG, RNN_THREADS, 0, stream>>>(xB, wx0B, wh0B, wx1B, wh1B,
                                             bh0, bh1, hidden,
                                             hB, hF, topsB, hid_out,
                                             bar, bar + 1);

  // 4) output projection: write-bound GEMM, Wout DMA'd to LDS (async, 2-buf)
  dim3 grid(V_ / PJ_NBLK, (S_ * B_) / PJ_MBLK);
  k_proj<<<grid, PJ_THREADS, 0, stream>>>(topsB, woutB, bout, out);
}
